// BidirLSTMLayer_42941083025997
// MI455X (gfx1250) — compile-verified, matrix operand order checked by reading, not run on hardware
//
#include <hip/hip_runtime.h>
#include <hip/hip_fp16.h>

typedef __attribute__((ext_vector_type(16))) _Float16 v16h;
typedef __attribute__((ext_vector_type(8)))  float    v8f;

#define T_STEPS 256
#define B_      128
#define IN_     512
#define H_      1024
#define KTOT    1536           // IN + H
#define G4H     4096           // 4*H
#define NWG_DIR 32             // workgroups per direction

// ---------------- workspace layout (bytes) ----------------
#define OFF_X    0
#define OFF_W    33554432ULL
#define OFF_BIAS 58720256ULL
#define OFF_H    58753024ULL
#define OFF_C    59801600ULL
#define OFF_CNT  60850176ULL

__device__ __forceinline__ float sigf(float x)   { return 1.0f / (1.0f + __expf(-x)); }
__device__ __forceinline__ float tanh_f(float x) { return 2.0f * sigf(2.0f * x) - 1.0f; }

// ---------------- prep kernels ----------------
__global__ void k_f32_to_f16(const float* __restrict__ src, _Float16* __restrict__ dst, int n) {
    for (int i = blockIdx.x * blockDim.x + threadIdx.x; i < n; i += gridDim.x * blockDim.x)
        dst[i] = (_Float16)src[i];
}

// wcat[dir][n][k], k<512 -> Wih[n][k], else Whh[n][k-512]
__global__ void k_build_wcat(const float* __restrict__ WihF, const float* __restrict__ WhhF,
                             const float* __restrict__ WihR, const float* __restrict__ WhhR,
                             _Float16* __restrict__ wcat) {
    const long long total = 2LL * G4H * KTOT;
    for (long long i = blockIdx.x * (long long)blockDim.x + threadIdx.x; i < total;
         i += (long long)gridDim.x * blockDim.x) {
        int dir = (int)(i / ((long long)G4H * KTOT));
        long long rem = i - (long long)dir * G4H * KTOT;
        int n = (int)(rem / KTOT);
        int k = (int)(rem - (long long)n * KTOT);
        const float* Wih = dir ? WihR : WihF;
        const float* Whh = dir ? WhhR : WhhF;
        float v = (k < IN_) ? Wih[(long long)n * IN_ + k]
                            : Whh[(long long)n * H_ + (k - IN_)];
        wcat[i] = (_Float16)v;
    }
}

__global__ void k_init_state(const float* __restrict__ h0f, const float* __restrict__ c0f,
                             const float* __restrict__ h0r, const float* __restrict__ c0r,
                             const float* __restrict__ bihf, const float* __restrict__ bhhf,
                             const float* __restrict__ bihr, const float* __restrict__ bhhr,
                             _Float16* __restrict__ hbuf, float* __restrict__ cbuf,
                             float* __restrict__ bias, int* __restrict__ cnt) {
    const int nstate = 2 * B_ * H_;
    for (int i = blockIdx.x * blockDim.x + threadIdx.x; i < nstate; i += gridDim.x * blockDim.x) {
        int dir = i / (B_ * H_);
        int j   = i - dir * (B_ * H_);
        hbuf[((long long)dir * 2 + 0) * B_ * H_ + j] = (_Float16)(dir ? h0r[j] : h0f[j]);
        cbuf[i] = dir ? c0r[j] : c0f[j];
        if (i < 2 * G4H) {
            int d = i / G4H, n = i - d * G4H;
            bias[i] = d ? (bihr[n] + bhhr[n]) : (bihf[n] + bhhf[n]);
        }
        if (i < 2) cnt[i] = 0;
    }
}

// Stage one 128x64 f16 A-block into LDS with the CDNA5 async-to-LDS path.
// 1024 chunks of 16B; 256 threads x 4 chunks. Tracked by ASYNCcnt.
__device__ __forceinline__ void stage_a(const _Float16* __restrict__ rowbase, int rstride,
                                        int k0, unsigned ldsbase, int tid) {
#pragma unroll
    for (int j = 0; j < 4; ++j) {
        const int c    = tid + j * 256;   // 0..1023
        const int row  = c >> 3;          // 0..127
        const int part = c & 7;           // 8 x (8 f16 = 16B) per 128B row
        const _Float16* g = rowbase + (size_t)row * rstride + k0 + part * 8;
        const unsigned  l = ldsbase + (unsigned)(row * 128 + part * 16);
        asm volatile("global_load_async_to_lds_b128 %0, %1, off"
                     :: "v"(l), "v"(g) : "memory");
    }
}

__device__ __forceinline__ void wait_async_lds() {
    asm volatile("s_wait_asynccnt 0x0" ::: "memory");
    __syncthreads();
}

// ---------------- persistent bidirectional LSTM scan ----------------
// grid = 64 blocks: dir = blk>>5, wg = blk&31 owns h-cols [32*wg, 32*wg+32)
// block = 256 threads = 8 wave32; wave w: gate = w>>1, N-half = w&1 -> 128x16 gate strip
__global__ __launch_bounds__(256) void lstm_scan_kernel(
    const _Float16* __restrict__ xf,    // [T][B][IN] f16
    const _Float16* __restrict__ wcat,  // [2][4096][1536] f16
    const float*    __restrict__ bias,  // [2][4096]
    _Float16*       __restrict__ hbuf,  // [2][2][B][H] f16 ping-pong
    float*          __restrict__ cbuf,  // [2][B][H]
    float*          __restrict__ out,   // [T][B][2H]
    float*          __restrict__ states,// [2 dirs][h;c][B][H]
    int*            __restrict__ cnt)   // [2] per-direction step counters
{
    const int dir  = blockIdx.x >> 5;
    const int wg   = blockIdx.x & 31;
    const int tid  = threadIdx.x;
    const int wave = tid >> 5;
    const int lane = tid & 31;
    const int gate  = wave >> 1;
    const int nhalf = wave & 1;
    const int ln = lane & 15;        // N (B-frag) / M-row (A-frag)
    const int lh = lane >> 4;        // K-half selector (+16)

    const int nbase = gate * H_ + wg * 32 + nhalf * 16;
    const _Float16* wrow = wcat + (size_t)dir * G4H * KTOT + (size_t)(nbase + ln) * KTOT;
    const float*    bdir = bias + dir * G4H;
    float*          cws  = cbuf + (size_t)dir * B_ * H_;

    __shared__ float ldsg[4][B_][32];                     // 64 KB gate exchange
    __shared__ __align__(128) _Float16 abuf[2][B_][64];   // 32 KB double-buffered A tile

    const unsigned abase[2] = {
        (unsigned)(unsigned long long)(void*)&abuf[0][0][0],
        (unsigned)(unsigned long long)(void*)&abuf[1][0][0] };

    for (int s = 0; s < T_STEPS; ++s) {
        const int t = dir ? (T_STEPS - 1 - s) : s;
        const _Float16* xbase = xf + (size_t)t * B_ * IN_;

        v8f acc[8] = {};   // 8 M-tiles x 16 cols, f32 accum

        // ======== x-portion (K = 0..511): independent of h, hides grid-sync latency
        stage_a(xbase, IN_, 0, abase[0], tid);
        wait_async_lds();
        int buf = 0;
        for (int k0 = 0; k0 < IN_; k0 += 64) {
            if (k0 + 64 < IN_) stage_a(xbase, IN_, k0 + 64, abase[buf ^ 1], tid);
#pragma unroll
            for (int kk = 0; kk < 64; kk += 32) {
                v16h bf = *(const v16h*)(wrow + k0 + kk + lh * 16);
#pragma unroll
                for (int mt = 0; mt < 8; ++mt) {
                    v16h af = *(const v16h*)(&abuf[buf][mt * 16 + ln][kk + lh * 16]);
                    acc[mt] = __builtin_amdgcn_wmma_f32_16x16x32_f16(
                        false, af, false, bf, (short)0, acc[mt], false, false);
                }
            }
            if (k0 + 64 < IN_) { wait_async_lds(); buf ^= 1; }
        }

        // ======== wait for h(s-1) from all 32 WGs of this direction
        if (s > 0 && tid == 0) {
            volatile int* vc = cnt + dir;
            const int target = NWG_DIR * s;
            while (*vc < target) __builtin_amdgcn_s_sleep(2);
            __threadfence();
        }
        __syncthreads();   // also fences x-stage LDS reuse below

        // ======== h-portion (K = 512..1535)
        const _Float16* hb = hbuf + (size_t)(dir * 2 + (s & 1)) * B_ * H_;
        stage_a(hb, H_, 0, abase[0], tid);
        wait_async_lds();
        buf = 0;
        for (int k0 = 0; k0 < H_; k0 += 64) {
            if (k0 + 64 < H_) stage_a(hb, H_, k0 + 64, abase[buf ^ 1], tid);
#pragma unroll
            for (int kk = 0; kk < 64; kk += 32) {
                v16h bf = *(const v16h*)(wrow + IN_ + k0 + kk + lh * 16);
#pragma unroll
                for (int mt = 0; mt < 8; ++mt) {
                    v16h af = *(const v16h*)(&abuf[buf][mt * 16 + ln][kk + lh * 16]);
                    acc[mt] = __builtin_amdgcn_wmma_f32_16x16x32_f16(
                        false, af, false, bf, (short)0, acc[mt], false, false);
                }
            }
            if (k0 + 64 < H_) { wait_async_lds(); buf ^= 1; }
        }

        // ======== spill gate strips to LDS so i,f,g,o meet in one thread
        // C layout: VGPR r holds M = r + 8*(lane>=16), N = lane&15
        const int colw = nhalf * 16 + ln;
#pragma unroll
        for (int mt = 0; mt < 8; ++mt)
#pragma unroll
            for (int r = 0; r < 8; ++r)
                ldsg[gate][mt * 16 + lh * 8 + r][colw] = acc[mt][r];
        __syncthreads();

        // ======== elementwise LSTM cell update: 128x32 elements, 16/thread
        _Float16* hnext = hbuf + (size_t)(dir * 2 + ((s + 1) & 1)) * B_ * H_;
#pragma unroll
        for (int j = 0; j < 16; ++j) {
            const int e   = tid + 256 * j;
            const int row = e >> 5;
            const int col = e & 31;
            const int hc  = wg * 32 + col;
            float iv = ldsg[0][row][col] + bdir[0 * H_ + hc];
            float fv = ldsg[1][row][col] + bdir[1 * H_ + hc];
            float gv = ldsg[2][row][col] + bdir[2 * H_ + hc];
            float ov = ldsg[3][row][col] + bdir[3 * H_ + hc];
            float co = cws[row * H_ + hc];
            float cn = sigf(fv) * co + sigf(iv) * tanh_f(gv);
            float hv = sigf(ov) * tanh_f(cn);
            cws[row * H_ + hc] = cn;
            hnext[(size_t)row * H_ + hc] = (_Float16)hv;
            out[(size_t)t * B_ * 2 * H_ + (size_t)row * 2 * H_ + (size_t)dir * H_ + hc] = hv;
            if (s == T_STEPS - 1) {
                // dir 0 -> "hidden" = [hf, cf]; dir 1 -> "cell" = [hr, cr]
                states[(size_t)dir * 2 * B_ * H_ + 0 * B_ * H_ + (size_t)row * H_ + hc] = hv;
                states[(size_t)dir * 2 * B_ * H_ + 1 * B_ * H_ + (size_t)row * H_ + hc] = cn;
            }
        }

        // ======== publish h(s)
        __threadfence();
        __syncthreads();
        if (tid == 0) atomicAdd(&cnt[dir], 1);
    }
}

extern "C" void kernel_launch(void* const* d_in, const int* in_sizes, int n_in,
                              void* d_out, int out_size, void* d_ws, size_t ws_size,
                              hipStream_t stream) {
    (void)in_sizes; (void)n_in; (void)out_size; (void)ws_size;
    const float* x     = (const float*)d_in[0];
    const float* h0f   = (const float*)d_in[1];
    const float* c0f   = (const float*)d_in[2];
    const float* h0r   = (const float*)d_in[3];
    const float* c0r   = (const float*)d_in[4];
    const float* WihF  = (const float*)d_in[5];
    const float* WhhF  = (const float*)d_in[6];
    const float* bihf  = (const float*)d_in[7];
    const float* bhhf  = (const float*)d_in[8];
    const float* WihR  = (const float*)d_in[9];
    const float* WhhR  = (const float*)d_in[10];
    const float* bihr  = (const float*)d_in[11];
    const float* bhhr  = (const float*)d_in[12];

    char* ws = (char*)d_ws;
    _Float16* xf16 = (_Float16*)(ws + OFF_X);
    _Float16* wcat = (_Float16*)(ws + OFF_W);
    float*    bias = (float*)   (ws + OFF_BIAS);
    _Float16* hbuf = (_Float16*)(ws + OFF_H);
    float*    cbuf = (float*)   (ws + OFF_C);
    int*      cnt  = (int*)     (ws + OFF_CNT);

    float* out    = (float*)d_out;
    float* states = out + (size_t)T_STEPS * B_ * 2 * H_;

    k_f32_to_f16<<<4096, 256, 0, stream>>>(x, xf16, T_STEPS * B_ * IN_);
    k_build_wcat<<<4096, 256, 0, stream>>>(WihF, WhhF, WihR, WhhR, wcat);
    k_init_state<<<1024, 256, 0, stream>>>(h0f, c0f, h0r, c0r,
                                           bihf, bhhf, bihr, bhhr,
                                           hbuf, cbuf, bias, cnt);
    lstm_scan_kernel<<<2 * NWG_DIR, 256, 0, stream>>>(
        xf16, wcat, bias, hbuf, cbuf, out, states, cnt);
}
